// RNN_6751688589707
// MI455X (gfx1250) — matrix operand order repaired and checked by reading, not compiled, not run on hardware
//
#include <hip/hip_runtime.h>

// Problem constants: x:(N,T,D) f32, h0:(N,H), Wx:(D,H), Wh:(H,H), b:(H)
#define N_ 128
#define T_ 512
#define D_ 1024
#define H_ 1024
#define SCAN_BLOCKS 64        // persistent scan: 64 blocks x 128 threads

typedef __attribute__((ext_vector_type(16))) __bf16 bf16x16;
typedef __attribute__((ext_vector_type(2)))  __bf16 bf16x2;
typedef __attribute__((ext_vector_type(8)))  float  floatx8;

__device__ __forceinline__ unsigned short f32_to_bf16_bits(float f) {
  __bf16 h = (__bf16)f;
  return __builtin_bit_cast(unsigned short, h);
}
__device__ __forceinline__ unsigned pack_bf16x2(float lo, float hi) {
  bf16x2 p;
  p.x = (__bf16)lo;
  p.y = (__bf16)hi;
  return __builtin_bit_cast(unsigned, p);
}
__device__ __forceinline__ float fast_tanhf(float xv) {
#if __has_builtin(__builtin_amdgcn_tanhf)
  return __builtin_amdgcn_tanhf(xv);
#else
  const float e = __expf(2.0f * xv);
  return 1.0f - 2.0f / (e + 1.0f);
#endif
}

union Frag16 {
  bf16x16 v;
  unsigned u[8];
  uint4    q[2];
};

// ---------------------------------------------------------------------------
// Prep: WhT[n*H + k] = bf16(Wh[k,n]), WxT[n*D + k] = bf16(Wx[k,n]), h0 -> bf16,
// and zero the grid-barrier counter (determinism across replays).
// ---------------------------------------------------------------------------
__global__ void rnn_prep_kernel(const float* __restrict__ Wh,
                                const float* __restrict__ Wx,
                                const float* __restrict__ h0,
                                unsigned short* __restrict__ WhT,
                                unsigned short* __restrict__ WxT,
                                unsigned short* __restrict__ h0b,
                                unsigned* __restrict__ barrier_cnt) {
  if (blockIdx.x == 0 && threadIdx.x == 0) *barrier_cnt = 0u;
  int i = blockIdx.x * blockDim.x + threadIdx.x;
  const int nWh = H_ * H_;
  const int nWx = D_ * H_;
  const int nH0 = N_ * H_;
  for (; i < nWh + nWx + nH0; i += gridDim.x * blockDim.x) {
    if (i < nWh) {
      const int k = i >> 10, n = i & (H_ - 1);
      WhT[(size_t)n * H_ + k] = f32_to_bf16_bits(Wh[(size_t)k * H_ + n]);
    } else if (i < nWh + nWx) {
      const int j = i - nWh;
      const int k = j >> 10, n = j & (H_ - 1);
      WxT[(size_t)n * D_ + k] = f32_to_bf16_bits(Wx[(size_t)k * H_ + n]);
    } else {
      const int j = i - nWh - nWx;
      h0b[j] = f32_to_bf16_bits(h0[j]);
    }
  }
}

// ---------------------------------------------------------------------------
// Phase 1 (fully parallel): preact = x @ Wx + b  written into d_out (f32).
// Grid = (H/64, N*T/64) = (16, 1024); block 256 thr; wave = 16x32 strip.
// ---------------------------------------------------------------------------
__global__ __launch_bounds__(256) void rnn_xproj_kernel(
    const float*          __restrict__ x,      // (N*T, D) f32
    const unsigned short* __restrict__ WxT,    // (H cols) x (D), bf16 bits
    const float*          __restrict__ bias,   // (H)
    float*                __restrict__ out) {  // (N*T, H) f32 pre-activations
  constexpr int P = 40;
  __shared__ __align__(16) unsigned short As[2][64][P];
  __shared__ __align__(16) unsigned short Bs[2][64][P];

  const int tid  = threadIdx.x;
  const int lane = tid & 31;
  const int wave = tid >> 5;
  const int half = lane >> 4;
  const int l16  = lane & 15;

  const int r0 = blockIdx.y * 64;
  const int n0 = blockIdx.x * 64;
  const int wm = wave & 3;
  const int wn = wave >> 2;

  const int srow = tid >> 2;
  const int sk8  = (tid & 3) * 8;

  const float*          xrow  = x + (size_t)(r0 + srow) * D_ + sk8;
  const unsigned short* wtrow = WxT + (size_t)(n0 + srow) * D_ + sk8;

  floatx8 c0 = {}, c1 = {};
  uint4 aReg, bReg;

  auto loadSlab = [&](int s) {
    const int ks = s * 32;
    bReg = *(const uint4*)(wtrow + ks);
    const float4 f0 = *(const float4*)(xrow + ks);
    const float4 f1 = *(const float4*)(xrow + ks + 4);
    aReg.x = pack_bf16x2(f0.x, f0.y);
    aReg.y = pack_bf16x2(f0.z, f0.w);
    aReg.z = pack_bf16x2(f1.x, f1.y);
    aReg.w = pack_bf16x2(f1.z, f1.w);
  };

  loadSlab(0);
  int buf = 0;
  for (int s = 0; s < D_ / 32; ++s) {
    *(uint4*)&As[buf][srow][sk8] = aReg;
    *(uint4*)&Bs[buf][srow][sk8] = bReg;
    __syncthreads();
    if (s + 1 < D_ / 32) loadSlab(s + 1);

    Frag16 a;
    const int arow = wm * 16 + l16;
    a.q[0] = *(const uint4*)&As[buf][arow][8 * half];
    a.q[1] = *(const uint4*)&As[buf][arow][16 + 8 * half];

    Frag16 fb0, fb1;
    const int brow = wn * 32 + l16;
    fb0.q[0] = *(const uint4*)&Bs[buf][brow][8 * half];
    fb0.q[1] = *(const uint4*)&Bs[buf][brow][16 + 8 * half];
    fb1.q[0] = *(const uint4*)&Bs[buf][brow + 16][8 * half];
    fb1.q[1] = *(const uint4*)&Bs[buf][brow + 16][16 + 8 * half];

    c0 = __builtin_amdgcn_wmma_f32_16x16x32_bf16(false, a.v, false, fb0.v,
                                                 (short)0, c0, false, false);
    c1 = __builtin_amdgcn_wmma_f32_16x16x32_bf16(false, a.v, false, fb1.v,
                                                 (short)0, c1, false, false);
    buf ^= 1;
  }

  const int rg_base = r0 + wm * 16 + 8 * half;
  const int nc0 = n0 + wn * 32 + l16;
  const int nc1 = nc0 + 16;
  const float bb0 = bias[nc0];
  const float bb1 = bias[nc1];
#pragma unroll
  for (int r = 0; r < 8; ++r) {
    const size_t row = (size_t)(rg_base + r);
    out[row * H_ + nc0] = c0[r] + bb0;
    out[row * H_ + nc1] = c1[r] + bb1;
  }
}

// ---------------------------------------------------------------------------
// Phase 2: persistent scan kernel. 64 blocks x 128 threads (4 waves).
// Block owns a 64-row x 32-col tile; its 32-column Wh panel (64 KB bf16)
// stays resident in LDS for all T steps (XOR-swizzled, conflict-free reads).
// Per step: 32 K-slabs, A fragments straight from global h_prev (L2-hot),
// 2 WMMA per wave per slab, no intra-step __syncthreads. Steps separated by
// a device-wide atomic arrive/spin barrier.
// ---------------------------------------------------------------------------
__global__ __launch_bounds__(128) void rnn_scan_kernel(
    const unsigned short* __restrict__ WhT,    // (H cols) x (H), bf16 bits
    unsigned short*       __restrict__ hb0,    // (N,H) bf16: h at even t
    unsigned short*       __restrict__ hb1,    // (N,H) bf16: h at odd t
    float*                __restrict__ out,    // (N,T,H) f32: preact in, h out
    unsigned*             __restrict__ barrier_cnt) {
  __shared__ __align__(16) unsigned short Bp[32 * 1024];   // 64 KB Wh panel

  const int tid  = threadIdx.x;
  const int lane = tid & 31;
  const int wave = tid >> 5;                   // 0..3 -> M tile
  const int half = lane >> 4;
  const int l16  = lane & 15;

  const int bid = blockIdx.x;
  const int n0  = (bid & 31) * 32;             // hidden-col base (0..992)
  const int m0  = (bid >> 5) * 64;             // batch-row base (0 or 64)

  // ---- load this block's Wh panel once, XOR-swizzled at 16B granularity ----
  // chunk (col c, q) lives at ushort offset c*1024 + ((q ^ (c & 15)) * 8)
  for (int i = 0; i < 32; ++i) {               // i == column index 0..31
    const int q = tid;                         // hope: 128 chunks? tid 0..127
    const uint4 w = *(const uint4*)&WhT[(size_t)(n0 + i) * H_ + q * 8];
    *(uint4*)&Bp[i * 1024 + ((q ^ (i & 15)) * 8)] = w;
  }
  __syncthreads();

  const int arow = m0 + wave * 16 + l16;       // A-fragment row for this lane
  const int mg_base = m0 + wave * 16 + 8 * half;
  const int nc0 = n0 + l16;
  const int nc1 = nc0 + 16;
  const unsigned swzrow0 = (unsigned)(l16 * 1024);
  const unsigned swzrow1 = (unsigned)((l16 + 16) * 1024);

  for (int t = 0; t < T_; ++t) {
    const unsigned short* hp = (t & 1) ? hb1 : hb0;
    unsigned short*       hn = (t & 1) ? hb0 : hb1;
    const unsigned short* aptr = hp + (size_t)arow * H_;

    floatx8 c0 = {}, c1 = {};
#pragma unroll 4
    for (int s = 0; s < H_ / 32; ++s) {
      Frag16 a;
      a.q[0] = *(const uint4*)(aptr + 32 * s + 8 * half);
      a.q[1] = *(const uint4*)(aptr + 32 * s + 16 + 8 * half);

      const int qb = 4 * s + half;             // 16B-chunk index base
      const unsigned off0 = (unsigned)((qb ^ l16) * 8);
      const unsigned off1 = (unsigned)(((qb + 2) ^ l16) * 8);
      Frag16 fb0, fb1;
      fb0.q[0] = *(const uint4*)&Bp[swzrow0 + off0];
      fb0.q[1] = *(const uint4*)&Bp[swzrow0 + off1];
      fb1.q[0] = *(const uint4*)&Bp[swzrow1 + off0];
      fb1.q[1] = *(const uint4*)&Bp[swzrow1 + off1];

      c0 = __builtin_amdgcn_wmma_f32_16x16x32_bf16(false, a.v, false, fb0.v,
                                                   (short)0, c0, false, false);
      c1 = __builtin_amdgcn_wmma_f32_16x16x32_bf16(false, a.v, false, fb1.v,
                                                   (short)0, c1, false, false);
    }

    // epilogue: h = tanh(hWh + preact); overwrite out tile, emit bf16 h_next
#pragma unroll
    for (int r = 0; r < 8; ++r) {
      const int mg = mg_base + r;
      const size_t o0 = ((size_t)mg * T_ + t) * H_ + nc0;
      const size_t o1 = ((size_t)mg * T_ + t) * H_ + nc1;
      const float v0 = fast_tanhf(c0[r] + out[o0]);
      const float v1 = fast_tanhf(c1[r] + out[o1]);
      out[o0] = v0;
      out[o1] = v1;
      hn[mg * H_ + nc0] = f32_to_bf16_bits(v0);
      hn[mg * H_ + nc1] = f32_to_bf16_bits(v1);
    }

    // ---- device-wide step barrier (arrive + spin) ----
    __threadfence();
    __syncthreads();
    if (tid == 0) {
      __hip_atomic_fetch_add(barrier_cnt, 1u, __ATOMIC_RELEASE,
                             __HIP_MEMORY_SCOPE_AGENT);
      const unsigned target = (unsigned)SCAN_BLOCKS * (unsigned)(t + 1);
      while (__hip_atomic_load(barrier_cnt, __ATOMIC_ACQUIRE,
                               __HIP_MEMORY_SCOPE_AGENT) < target) {
        __builtin_amdgcn_s_sleep(1);
      }
    }
    __syncthreads();
    __threadfence();
  }
}

// ---------------------------------------------------------------------------
extern "C" void kernel_launch(void* const* d_in, const int* in_sizes, int n_in,
                              void* d_out, int out_size, void* d_ws, size_t ws_size,
                              hipStream_t stream) {
  (void)in_sizes; (void)n_in; (void)out_size; (void)ws_size;

  const float* x   = (const float*)d_in[0];  // (N,T,D)
  const float* h0  = (const float*)d_in[1];  // (N,H)
  const float* Wx  = (const float*)d_in[2];  // (D,H)
  const float* Wh  = (const float*)d_in[3];  // (H,H)
  const float* b   = (const float*)d_in[4];  // (H)
  float*       out = (float*)d_out;          // (N,T,H)

  // Workspace: WhT (2MB) | WxT (2MB) | hb0 | hb1 (256KB each) | barrier cnt
  unsigned short* WhT = (unsigned short*)d_ws;
  unsigned short* WxT = WhT + (size_t)H_ * H_;
  unsigned short* hb0 = WxT + (size_t)D_ * H_;
  unsigned short* hb1 = hb0 + (size_t)N_ * H_;
  unsigned*       cnt = (unsigned*)(hb1 + (size_t)N_ * H_);

  rnn_prep_kernel<<<2048, 256, 0, stream>>>(Wh, Wx, h0, WhT, WxT, hb0, cnt);

  dim3 gproj(H_ / 64, (N_ * T_) / 64);       // (16, 1024)
  rnn_xproj_kernel<<<gproj, 256, 0, stream>>>(x, WxT, b, out);

  rnn_scan_kernel<<<SCAN_BLOCKS, 128, 0, stream>>>(WhT, hb0, hb1, out, cnt);
}
